// SingleStageDetector_62964220559551
// MI455X (gfx1250) — matrix operand order repaired and checked by reading, not compiled, not run on hardware
//
#include <hip/hip_runtime.h>
#include <math.h>

// ---------------- problem constants ----------------
constexpr int kB   = 16;
constexpr int kIN  = 512;
constexpr int kHID = 256;
constexpr int kA   = 9;
constexpr int kOUT = 65;      // C + 5*A = 20 + 45
constexpr int kH   = 64;
constexpr int kW   = 64;
constexpr int kHW  = kH * kW;           // 4096
constexpr int kP   = kB * kHW;          // 65536 pixels
constexpr int kN   = 50;
constexpr int kM   = 256;
constexpr int kK   = 2000;              // K_NMS
constexpr float kIouThr = 0.5f;
constexpr int kPA  = kA * kHW;          // 36864 anchors per image

// ---------------- WMMA vector types ----------------
typedef __attribute__((ext_vector_type(16))) __bf16 v16bf;
typedef __attribute__((ext_vector_type(8)))  float  v8f;

__device__ __forceinline__ unsigned short bf_bits(float x) {
    unsigned u = __builtin_bit_cast(unsigned, x);
    unsigned r = u + 0x7fffu + ((u >> 16) & 1u);   // round-to-nearest-even
    return (unsigned short)(r >> 16);
}
__device__ __forceinline__ __bf16 f2bf(float x) {
    return __builtin_bit_cast(__bf16, bf_bits(x));
}
__device__ __forceinline__ __bf16 us2bf(unsigned short s) {
    return __builtin_bit_cast(__bf16, s);
}
__device__ __forceinline__ unsigned pack_bf2(float lo, float hi) {
    return (unsigned)bf_bits(lo) | ((unsigned)bf_bits(hi) << 16);
}
__device__ __forceinline__ float sigm(float x) {
    return 1.0f / (1.0f + expf(-x));
}

// =====================================================================
// Kernel 1: h = leakyrelu(features^T @ W1^T + b1), stored bf16 (P, HID)
// block = 256 threads = 8 waves. Block tile: 64 pixels x 256 outputs
// (features read exactly ONCE from HBM). Wave: 16 pixels x 128 outputs
// = 8 accumulators; A-fragment loaded once per K-step, reused 8x.
// W1 K-slice staged in LDS pre-swizzled into B-fragment layout.
// =====================================================================
__global__ void __launch_bounds__(256)
k_gemm1_wmma(const float* __restrict__ feat,   // (B, IN, H, W)
             const float* __restrict__ W1,     // (HID, IN)
             const float* __restrict__ b1,     // (HID)
             unsigned short* __restrict__ h_ws)// (P, HID) bf16
{
    // [o-tile 0..15][lane 0..31][elem 0..15] : lane-private contiguous 32B
    __shared__ unsigned short w1s[16][32][16];   // 16 KB

    const int lane = threadIdx.x & 31;
    const int wave = threadIdx.x >> 5;
    const int pm   = wave & 3;          // pixel sub-tile 0..3
    const int oh   = wave >> 2;         // output half 0..1 (128 outputs each)
    const int p0   = blockIdx.x * 64 + pm * 16;

    const int m   = lane & 15;
    const int hi  = lane >> 4;          // 0 or 1
    const int kbA = hi * 16;
    const int p   = p0 + m;
    const int b   = p >> 12;            // tile never straddles images
    const int q   = p & 4095;

    const float* fbase = feat + (size_t)b * kIN * kHW + q;  // channel stride kHW

    v8f acc[8] = {};

    for (int c0 = 0; c0 < kIN; c0 += 32) {
        // ---- stage W1[:, c0:c0+32] into LDS, swizzled to B-frag layout ----
        {
            const int o = threadIdx.x;                   // 0..255
            const int t = o >> 4;                        // o-tile
            const int n = o & 15;                        // column within tile
            const float* wr = W1 + (size_t)o * kIN + c0; // 32 contiguous f32
#pragma unroll
            for (int k = 0; k < 32; k += 2) {
                const int bl = (k < 16) ? n : (n + 16);  // B-frag lane
                const int j  = k & 15;                   // element (even)
                *(unsigned*)&w1s[t][bl][j] = pack_bf2(wr[k], wr[k + 1]);
            }
        }

        if (c0 + 32 < kIN)
            __builtin_prefetch(fbase + (size_t)(c0 + 32) * kHW, 0, 1);

        // ---- A fragment: 16 pixels x 32 channels (strided global reads) ----
        v16bf afrag;
#pragma unroll
        for (int j = 0; j < 16; ++j) {
            const int k = c0 + kbA + j;
            afrag[j] = f2bf(fbase[(size_t)k * kHW]);
        }

        __syncthreads();   // staging visible

        // ---- 8 WMMAs reusing afrag, B-frags from LDS (32B/lane reads) ----
#pragma unroll
        for (int it = 0; it < 8; ++it) {
            const int tt = oh * 8 + it;
            const unsigned short* brow = &w1s[tt][lane][0];
            v16bf bfrag;
#pragma unroll
            for (int j = 0; j < 16; ++j) bfrag[j] = us2bf(brow[j]);
            acc[it] = __builtin_amdgcn_wmma_f32_16x16x32_bf16(
                          false, afrag, false, bfrag, (short)0, acc[it], false, false);
        }

        __syncthreads();   // before restaging
    }

    // ---- epilogue: bias + leaky relu, store bf16 ----
#pragma unroll
    for (int it = 0; it < 8; ++it) {
        const int o0 = (oh * 8 + it) * 16;
        const float bias = b1[o0 + m];
#pragma unroll
        for (int r = 0; r < 8; ++r) {
            const int row = p0 + r + hi * 8;
            float v = acc[it][r] + bias;
            v = v > 0.0f ? v : 0.01f * v;
            h_ws[(size_t)row * kHID + (o0 + m)] = bf_bits(v);
        }
    }
}

// =====================================================================
// Kernel 2: out2 = h @ W2^T + b2, fused detector head.
// block = 256 threads = 8 waves, each wave 16 pixels x 16 outputs.
// A-tile (128 pix x 32 ch bf16, 8 KB) staged via CDNA5 async copy
// global_load_async_to_lds_b128 + s_wait_asynccnt.
// =====================================================================
__global__ void __launch_bounds__(256)
k_gemm2_head(const unsigned short* __restrict__ h_ws, // (P, HID) bf16
             const float* __restrict__ W2,            // (OUT, HID)
             const float* __restrict__ b2,            // (OUT)
             float* __restrict__ conf_ws,             // (B, A, HW)
             float* __restrict__ offs_ws)             // (B, A, HW, 4)
{
    __shared__ unsigned short htile[128 * 32];        // 8 KB, row-major [pix][k]

    const int lane = threadIdx.x & 31;
    const int wave = threadIdx.x >> 5;
    const int p0   = blockIdx.x * 128 + wave * 16;
    const int o0   = blockIdx.y * 16;

    const int m   = lane & 15;
    const int hi  = lane >> 4;
    const int kbA = hi * 8;
    const int kbB = hi * 16;
    const int o   = o0 + m;            // this lane's output column

    const size_t hbase = (size_t)blockIdx.x * 128 * kHID;

    v8f acc = {};
    for (int c0 = 0; c0 < kHID; c0 += 32) {
        // ---- async stage: 128 rows x 64B = 512 x 16B chunks, 2/thread ----
#pragma unroll
        for (int t = 0; t < 2; ++t) {
            const int ci  = threadIdx.x + t * 256;
            const int row = ci >> 2;
            const int cc  = (ci & 3) * 8;               // bf16 column chunk
            const unsigned short* g = h_ws + hbase + (size_t)row * kHID + c0 + cc;
            const unsigned long long ga = (unsigned long long)(size_t)g;
            const unsigned la = (unsigned)(size_t)&htile[row * 32 + cc];
            asm volatile("global_load_async_to_lds_b128 %0, %1, off"
                         :: "v"(la), "v"(ga) : "memory");
        }
        asm volatile("s_wait_asynccnt 0" ::: "memory");
        __syncthreads();

        // ---- fragments ----
        v16bf afrag, bfrag;
        const unsigned short* arow = &htile[(wave * 16 + m) * 32 + kbA];
#pragma unroll
        for (int j = 0; j < 8; ++j) {
            afrag[j]     = us2bf(arow[j]);
            afrag[j + 8] = us2bf(arow[j + 16]);
        }
#pragma unroll
        for (int j = 0; j < 16; ++j) {
            const float wv = (o < kOUT) ? W2[(size_t)o * kHID + c0 + kbA + (j < 8 ? j : j + 8)] : 0.0f;
            bfrag[j] = f2bf(wv);
        }
        acc = __builtin_amdgcn_wmma_f32_16x16x32_bf16(
                  false, afrag, false, bfrag, (short)0, acc, false, false);

        __syncthreads();   // before restaging
    }

    const float bias = (o < kOUT) ? b2[o] : 0.0f;
#pragma unroll
    for (int r = 0; r < 8; ++r) {
        const int p  = p0 + r + hi * 8;
        const int bb = p >> 12;
        const int q  = p & 4095;
        const float v = acc[r] + bias;
        if (o < kA) {                                      // confidence channels
            conf_ws[((size_t)bb * kA + o) * kHW + q] = sigm(v);
        } else if (o < 5 * kA) {                           // offset channels
            const int rel  = o - kA;
            const int ai   = rel >> 2;
            const int comp = rel & 3;
            const float vv = (comp < 2) ? (sigm(v) - 0.5f) : v;
            offs_ws[(((size_t)bb * kA + ai) * kHW + q) * 4 + comp] = vv;
        }
        // class channels 45..64 unused by reference outputs
    }
}

// =====================================================================
// Kernel 3: proposals from anchors + offsets
// =====================================================================
__global__ void k_proposals(const float* __restrict__ offs_ws,
                            const float* __restrict__ anc,   // (A,2)
                            const float* __restrict__ grid,  // (B,H,W,2)
                            float* __restrict__ props_ws)    // (B, A*HW, 4)
{
    const int idx = blockIdx.x * blockDim.x + threadIdx.x;
    if (idx >= kB * kA * kHW) return;
    const int q = idx & 4095;
    const int a = (idx >> 12) % kA;
    const int b = idx / (kA * kHW);

    const float gx = grid[((size_t)b * kHW + q) * 2 + 0];
    const float gy = grid[((size_t)b * kHW + q) * 2 + 1];
    const float aw = anc[a * 2 + 0];
    const float ah = anc[a * 2 + 1];

    const float* of = offs_ws + (size_t)idx * 4;
    const float cx = gx + of[0];
    const float cy = gy + of[1];
    const float pw = aw * expf(of[2]);
    const float ph = ah * expf(of[3]);

    float* pr = props_ws + (size_t)idx * 4;
    pr[0] = cx - 0.5f * pw;
    pr[1] = cy - 0.5f * ph;
    pr[2] = cx + 0.5f * pw;
    pr[3] = cy + 0.5f * ph;
}

// =====================================================================
// Kernel 4: iou_mat (B, 36864, 50)
// =====================================================================
__global__ void k_ioumat(const float* __restrict__ props_ws,
                         const float* __restrict__ bboxes,   // (B, N, 5)
                         float* __restrict__ out_iou)
{
    const int t = blockIdx.x * blockDim.x + threadIdx.x;
    if (t >= kB * kPA) return;
    const int b = t / kPA;

    const float* p = props_ws + (size_t)t * 4;
    const float px1 = p[0], py1 = p[1], px2 = p[2], py2 = p[3];
    const float pa = (px2 - px1) * (py2 - py1);

    const float* gb = bboxes + (size_t)b * kN * 5;
    float* op = out_iou + (size_t)t * kN;
#pragma unroll 5
    for (int n = 0; n < kN; ++n) {
        const float gx1 = gb[n * 5 + 0], gy1 = gb[n * 5 + 1];
        const float gx2 = gb[n * 5 + 2], gy2 = gb[n * 5 + 3];
        const float tlx = fmaxf(px1, gx1), tly = fmaxf(py1, gy1);
        const float brx = fminf(px2, gx2), bry = fminf(py2, gy2);
        const float inter = (brx > tlx && bry > tly) ? (brx - tlx) * (bry - tly) : 0.0f;
        const float ga = (gx2 - gx1) * (gy2 - gy1);
        op[n] = inter / (pa + ga - inter);
    }
}

// =====================================================================
// Kernel 5: conf + bbox losses (single block reduction)
// =====================================================================
__global__ void __launch_bounds__(512)
k_loss(const float* __restrict__ conf_ws,
       const float* __restrict__ offs_ws,
       const float* __restrict__ GT,        // (M,4)
       const int* __restrict__ pos,
       const int* __restrict__ neg,
       float* __restrict__ out_loss)
{
    __shared__ float sc[512];
    __shared__ float sb[512];
    const int t = threadIdx.x;
    float cl = 0.0f, bl = 0.0f;
    if (t < kM) {
        const int pi = pos[t];
        const float c = conf_ws[pi];
        cl = (c - 1.0f) * (c - 1.0f);
#pragma unroll
        for (int j = 0; j < 4; ++j) {
            const float d = offs_ws[(size_t)pi * 4 + j] - GT[t * 4 + j];
            bl += d * d;
        }
    } else {
        const int ni = neg[t - kM];
        const float c = conf_ws[ni];
        cl = c * c;
    }
    sc[t] = cl; sb[t] = bl;
    __syncthreads();
    for (int s = 256; s > 0; s >>= 1) {
        if (t < s) { sc[t] += sc[t + s]; sb[t] += sb[t + s]; }
        __syncthreads();
    }
    if (t == 0) out_loss[0] = sc[0] / (2.0f * kM) + sb[0] / kM;
}

// =====================================================================
// Kernel 6: top-K selection on image 0 (iterative argmax, single block)
// =====================================================================
__global__ void __launch_bounds__(1024)
k_topk(const float* __restrict__ conf_ws,     // first 36864 entries = image 0
       float* __restrict__ scores_ws,
       int* __restrict__ topidx_ws,
       float* __restrict__ out_topidx)
{
    __shared__ float sv[1024];
    __shared__ int   si[1024];
    const int t = threadIdx.x;
    for (int i = t; i < kPA; i += 1024) scores_ws[i] = conf_ws[i];
    __syncthreads();

    for (int k = 0; k < kK; ++k) {
        float best = -INFINITY; int bi = kPA;
        for (int i = t; i < kPA; i += 1024) {
            const float v = scores_ws[i];
            if (v > best) { best = v; bi = i; }
        }
        sv[t] = best; si[t] = bi;
        __syncthreads();
        for (int s = 512; s > 0; s >>= 1) {
            if (t < s) {
                if (sv[t + s] > sv[t] ||
                    (sv[t + s] == sv[t] && si[t + s] < si[t])) {
                    sv[t] = sv[t + s]; si[t] = si[t + s];
                }
            }
            __syncthreads();
        }
        if (t == 0) {
            topidx_ws[k]  = si[0];
            out_topidx[k] = (float)si[0];
            scores_ws[si[0]] = -INFINITY;
        }
        __syncthreads();
    }
}

// =====================================================================
// Kernel 7a: gather top boxes; 7b: pairwise IoU 2000x2000
// =====================================================================
__global__ void k_gather_boxes(const float* __restrict__ props_ws,
                               const int* __restrict__ topidx_ws,
                               float* __restrict__ boxes0)
{
    const int t = blockIdx.x * blockDim.x + threadIdx.x;
    if (t >= kK) return;
    const int idx = topidx_ws[t];                  // image 0 = first 36864 rows
    const float* p = props_ws + (size_t)idx * 4;
    float* o = boxes0 + (size_t)t * 4;
    o[0] = p[0]; o[1] = p[1]; o[2] = p[2]; o[3] = p[3];
}

__global__ void k_pairiou(const float* __restrict__ boxes0,
                          float* __restrict__ pair)
{
    const int j = blockIdx.x * 16 + threadIdx.x;
    const int i = blockIdx.y * 16 + threadIdx.y;
    if (i >= kK || j >= kK) return;
    const float* a = boxes0 + (size_t)i * 4;
    const float* b = boxes0 + (size_t)j * 4;
    const float tlx = fmaxf(a[0], b[0]), tly = fmaxf(a[1], b[1]);
    const float brx = fminf(a[2], b[2]), bry = fminf(a[3], b[3]);
    const float inter = (brx > tlx && bry > tly) ? (brx - tlx) * (bry - tly) : 0.0f;
    const float pa = (a[2] - a[0]) * (a[3] - a[1]);
    const float ga = (b[2] - b[0]) * (b[3] - b[1]);
    pair[(size_t)i * kK + j] = inter / (pa + ga - inter);
}

// =====================================================================
// Kernel 8: greedy NMS (sequential over i, parallel over j)
// =====================================================================
__global__ void __launch_bounds__(1024)
k_nms(const float* __restrict__ pair, float* __restrict__ out_keep)
{
    __shared__ unsigned char keep[kK];
    const int t = threadIdx.x;
    for (int i = t; i < kK; i += 1024) keep[i] = 1;
    __syncthreads();
    for (int i = 0; i < kK; ++i) {
        if (keep[i]) {
            for (int j = i + 1 + t; j < kK; j += 1024) {
                if (pair[(size_t)i * kK + j] > kIouThr) keep[j] = 0;
            }
        }
        __syncthreads();
    }
    for (int i = t; i < kK; i += 1024) out_keep[i] = keep[i] ? 1.0f : 0.0f;
}

// =====================================================================
// launch
// =====================================================================
extern "C" void kernel_launch(void* const* d_in, const int* in_sizes, int n_in,
                              void* d_out, int out_size, void* d_ws, size_t ws_size,
                              hipStream_t stream) {
    (void)in_sizes; (void)n_in; (void)out_size; (void)ws_size;

    const float* feat   = (const float*)d_in[0];   // (B, IN, H, W)
    const float* W1     = (const float*)d_in[1];   // (HID, IN)
    const float* b1     = (const float*)d_in[2];   // (HID)
    const float* W2     = (const float*)d_in[3];   // (OUT, HID)
    const float* b2     = (const float*)d_in[4];   // (OUT)
    const float* anc    = (const float*)d_in[5];   // (A,2)
    const float* grid   = (const float*)d_in[6];   // (B,H,W,2)
    const float* bboxes = (const float*)d_in[7];   // (B,N,5)
    const float* GT     = (const float*)d_in[8];   // (M,4)
    const int*   pos    = (const int*)d_in[9];     // (M)
    const int*   neg    = (const int*)d_in[10];    // (M)

    // ---- workspace layout ----
    char* ws = (char*)d_ws;
    size_t off = 0;
    unsigned short* h_ws = (unsigned short*)(ws + off); off += (size_t)kP * kHID * 2;        // 32 MB
    float* conf_ws   = (float*)(ws + off); off += (size_t)kB * kA * kHW * 4;                 // 2.4 MB
    float* offs_ws   = (float*)(ws + off); off += (size_t)kB * kA * kHW * 4 * 4;             // 9.4 MB
    float* props_ws  = (float*)(ws + off); off += (size_t)kB * kA * kHW * 4 * 4;             // 9.4 MB
    float* scores_ws = (float*)(ws + off); off += (size_t)kPA * 4;                           // 144 KB
    int*   topidx_ws = (int*)(ws + off);   off += (size_t)kK * 4;
    float* boxes0_ws = (float*)(ws + off); off += (size_t)kK * 4 * 4;
    float* pair_ws   = (float*)(ws + off); off += (size_t)kK * kK * 4;                       // 16 MB

    // ---- output layout (tuple concatenated flat) ----
    float* out        = (float*)d_out;
    float* out_iou    = out;                                  // B*36864*50
    float* out_loss   = out + (size_t)kB * kPA * kN;          // 1
    float* out_topidx = out_loss + 1;                         // 2000
    float* out_keep   = out_topidx + kK;                      // 2000

    // 1) GEMM1 (WMMA bf16): 1024 blocks, each 64 pixels x all 256 outputs
    k_gemm1_wmma<<<dim3(kP / 64), 256, 0, stream>>>(feat, W1, b1, h_ws);

    // 2) GEMM2 (WMMA bf16, async-to-LDS A tiles) + fused head: 512 x 5 blocks
    k_gemm2_head<<<dim3(kP / 128, 5), 256, 0, stream>>>(h_ws, W2, b2, conf_ws, offs_ws);

    // 3) proposals
    const int nProp = kB * kA * kHW;
    k_proposals<<<(nProp + 255) / 256, 256, 0, stream>>>(offs_ws, anc, grid, props_ws);

    // 4) IoU matrix (largest output stream)
    k_ioumat<<<(nProp + 255) / 256, 256, 0, stream>>>(props_ws, bboxes, out_iou);

    // 5) losses
    k_loss<<<1, 512, 0, stream>>>(conf_ws, offs_ws, GT, pos, neg, out_loss);

    // 6) top-K on image 0
    k_topk<<<1, 1024, 0, stream>>>(conf_ws, scores_ws, topidx_ws, out_topidx);

    // 7) gather + pairwise IoU
    k_gather_boxes<<<(kK + 255) / 256, 256, 0, stream>>>(props_ws, topidx_ws, boxes0_ws);
    k_pairiou<<<dim3((kK + 15) / 16, (kK + 15) / 16), dim3(16, 16), 0, stream>>>(boxes0_ws, pair_ws);

    // 8) greedy NMS
    k_nms<<<1, 1024, 0, stream>>>(pair_ws, out_keep);
}
